// MixedChunkAttention_44770739093837
// MI455X (gfx1250) — matrix operand order repaired and checked by reading, not compiled, not run on hardware
//
#include <hip/hip_runtime.h>
#include <hip/hip_bf16.h>
#include <math.h>
#include <stdint.h>

typedef __bf16 bf16;
typedef __attribute__((ext_vector_type(16))) __bf16 v16bf;
typedef __attribute__((ext_vector_type(8)))  __bf16 v8bf;
typedef __attribute__((ext_vector_type(8)))  float   v8f;
typedef unsigned int u32x4 __attribute__((ext_vector_type(4)));
typedef int          i32x4 __attribute__((ext_vector_type(4)));
typedef int          i32x8 __attribute__((ext_vector_type(8)));

struct V8x2 { v8bf lo, hi; };

static __device__ __forceinline__ v16bf cat16(v8bf lo, v8bf hi) {
  V8x2 t; t.lo = lo; t.hi = hi;
  return __builtin_bit_cast(v16bf, t);
}

// A fragment: 16x32 bf16, row-major, K-contiguous (ld in elements).
// Lanes 0-15: M=lane, K {0..7} in V0..3, K {16..23} in V4..7.
// Lanes 16-31: M=lane-16, K {8..15}, {24..31}.
static __device__ __forceinline__ v16bf frag_a(const bf16* base, int ld, int lane) {
  const bf16* p = base + (lane & 15) * ld + ((lane >> 4) << 3);
  return cat16(*(const v8bf*)p, *(const v8bf*)(p + 16));
}

// B fragment: 32x16 bf16 taken from an N x K row-major buffer (row n = column n of B).
// Lane carries column n = lane&15, K {0..7},{16..23} (lanes 0-15) or {8..15},{24..31} (lanes 16-31).
static __device__ __forceinline__ v16bf frag_b(const bf16* base, int ld, int lane) {
  const bf16* p = base + (lane & 15) * ld + ((lane >> 4) << 3);
  return cat16(*(const v8bf*)p, *(const v8bf*)(p + 16));
}

static __device__ __forceinline__ v8f wmma_bf16(v16bf a, v16bf b, v8f c) {
  return __builtin_amdgcn_wmma_f32_16x16x32_bf16(false, a, false, b, (short)0, c, false, false);
}

static __device__ __forceinline__ float silu_f(float x) { return x / (1.f + expf(-x)); }

// TDM: load a 2D bf16 tile (tile0 elems x tile1 rows, row stride = stride0 elems)
// from global memory into LDS. Descriptor per CDNA5 ISA 8.3/8.4. Caller must be
// a single wave; completion via s_wait_tensorcnt 0.
static __device__ __forceinline__ void tdm_load_2d_bf16(const bf16* gsrc, uint32_t lds_off,
                                                        uint32_t tile0, uint32_t tile1,
                                                        uint32_t stride0) {
  const uint64_t ga = (uint64_t)(uintptr_t)gsrc;
  u32x4 g0;
  g0.x = 1u;                                   // count=1 (valid), user desc
  g0.y = lds_off;                              // lds_addr (bytes)
  g0.z = (uint32_t)ga;                         // global_addr[31:0]
  g0.w = (uint32_t)((ga >> 32) & 0x01FFFFFFu)  // global_addr[56:32]
         | (2u << 30);                         // type = 2 ("image")
  const uint32_t td0 = stride0;                // tensor_dim0 (elems)
  const uint32_t td1 = tile1;                  // tensor_dim1 (rows)
  const uint64_t st0 = (uint64_t)stride0;      // tensor_dim0_stride
  const uint64_t st1 = (uint64_t)stride0 * tile1;
  i32x8 g1;
  g1[0] = (int)(1u << 16);                               // wg_mask=0, data_size=1 (2B)
  g1[1] = (int)((td0 & 0xFFFFu) << 16);                  // abar=0 | tensor_dim0[15:0]
  g1[2] = (int)((td0 >> 16) | ((td1 & 0xFFFFu) << 16));  // dim0[31:16] | dim1[15:0]
  g1[3] = (int)((td1 >> 16) | (tile0 << 16));            // dim1[31:16] | tile_dim0
  g1[4] = (int)(tile1 & 0xFFFFu);                        // tile_dim1 | tile_dim2=0
  g1[5] = (int)(uint32_t)(st0 & 0xFFFFFFFFu);            // stride0[31:0]
  g1[6] = (int)((uint32_t)((st0 >> 32) & 0xFFFFu)
         | ((uint32_t)(st1 & 0xFFFFu) << 16));           // stride0[47:32] | stride1[15:0]
  g1[7] = (int)(uint32_t)((st1 >> 16) & 0xFFFFFFFFu);    // stride1[47:16]
  i32x4 gz = {0, 0, 0, 0};
#if __clang_major__ >= 23
  i32x8 gz8 = {0, 0, 0, 0, 0, 0, 0, 0};
  __builtin_amdgcn_tensor_load_to_lds(g0, g1, gz, gz, gz8, 0);
#else
  __builtin_amdgcn_tensor_load_to_lds(g0, g1, gz, gz, 0);
#endif
}

// ---------------- LayerNorm (row per block, 256 threads) ----------------
template <typename OutT>
__global__ void k_layernorm(const float* __restrict__ x, const float* __restrict__ w,
                            const float* __restrict__ b, OutT* __restrict__ out, int D) {
  const int row = blockIdx.x;
  const float* xr = x + (size_t)row * D;
  __shared__ float red[256];
  float s = 0.f;
  for (int i = threadIdx.x; i < D; i += 256) s += xr[i];
  red[threadIdx.x] = s; __syncthreads();
  for (int o = 128; o > 0; o >>= 1) { if (threadIdx.x < o) red[threadIdx.x] += red[threadIdx.x + o]; __syncthreads(); }
  const float mean = red[0] * (1.f / D);
  __syncthreads();
  float vs = 0.f;
  for (int i = threadIdx.x; i < D; i += 256) { float d = xr[i] - mean; vs += d * d; }
  red[threadIdx.x] = vs; __syncthreads();
  for (int o = 128; o > 0; o >>= 1) { if (threadIdx.x < o) red[threadIdx.x] += red[threadIdx.x + o]; __syncthreads(); }
  const float rstd = rsqrtf(red[0] * (1.f / D) + 1e-5f);
  OutT* orow = out + (size_t)row * D;
  for (int i = threadIdx.x; i < D; i += 256)
    orow[i] = (OutT)((xr[i] - mean) * rstd * w[i] + b[i]);
}

// ---------------- f32 -> bf16 weight conversion ----------------
__global__ void k_cvt_bf16(const float* __restrict__ in, bf16* __restrict__ out, int n) {
  const int i = blockIdx.x * 256 + threadIdx.x;
  if (i < n) out[i] = (bf16)in[i];
}

// ---------------- Generic WMMA GEMM: out = act(A @ W^T + bias) ----------------
// A: (M,K) bf16 row-major. W: (Nout,K) bf16 row-major. Block tile 64x128, 8 waves,
// each wave: 16 rows x 64 cols (four 16x16 f32 accumulators; A frag reused 4x).
template <bool SILU, typename OutT>
__global__ void k_gemm(const bf16* __restrict__ A, const bf16* __restrict__ W,
                       const float* __restrict__ bias, OutT* __restrict__ out,
                       int K, int Nout) {
  const int lane = threadIdx.x & 31;
  const int wid  = threadIdx.x >> 5;
  const int mBase = blockIdx.x * 64 + ((wid >> 1) << 4);
  const int nBase = blockIdx.y * 128 + ((wid & 1) << 6);
  const bf16* Arow = A + (size_t)mBase * K;
  const bf16* Wp0 = W + (size_t)(nBase +  0) * K;
  const bf16* Wp1 = W + (size_t)(nBase + 16) * K;
  const bf16* Wp2 = W + (size_t)(nBase + 32) * K;
  const bf16* Wp3 = W + (size_t)(nBase + 48) * K;
  v8f acc0 = {}, acc1 = {}, acc2 = {}, acc3 = {};
  for (int k0 = 0; k0 < K; k0 += 32) {
    __builtin_prefetch(Arow + k0 + 256, 0, 3);   // global_prefetch_b8: next K chunk
    v16bf a = frag_a(Arow + k0, K, lane);
    acc0 = wmma_bf16(a, frag_b(Wp0 + k0, K, lane), acc0);
    acc1 = wmma_bf16(a, frag_b(Wp1 + k0, K, lane), acc1);
    acc2 = wmma_bf16(a, frag_b(Wp2 + k0, K, lane), acc2);
    acc3 = wmma_bf16(a, frag_b(Wp3 + k0, K, lane), acc3);
  }
  const int mo = (lane >> 4) << 3;
  const int c  = nBase + (lane & 15);
  for (int r = 0; r < 8; ++r) {
    const size_t row = (size_t)(mBase + r + mo);
    float v0 = acc0[r] + bias[c +  0];
    float v1 = acc1[r] + bias[c + 16];
    float v2 = acc2[r] + bias[c + 32];
    float v3 = acc3[r] + bias[c + 48];
    if (SILU) { v0 = silu_f(v0); v1 = silu_f(v1); v2 = silu_f(v2); v3 = silu_f(v3); }
    out[row * Nout + c +  0] = (OutT)v0;
    out[row * Nout + c + 16] = (OutT)v1;
    out[row * Nout + c + 32] = (OutT)v2;
    out[row * Nout + c + 48] = (OutT)v3;
  }
}

// ---------------- heads = qk*gamma+beta per head, then RoPE ----------------
__global__ void k_heads_rope(const bf16* __restrict__ qk, const float* __restrict__ gamma,
                             const float* __restrict__ beta,
                             bf16* __restrict__ qq, bf16* __restrict__ lq,
                             bf16* __restrict__ qkk, bf16* __restrict__ lk,
                             int N, int QK) {
  const int row  = blockIdx.x;            // b*N + n
  const int n    = row & (N - 1);         // absolute position
  const int j    = threadIdx.x;           // 0..127
  const int half = QK >> 1;               // 128
  const bf16* q = qk + (size_t)row * QK;
  const float a1 = (float)q[j];
  const float a2 = (float)q[j + half];
  // 10000^(-j/half) = exp2(-j * log2(10000)/half)
  const float inv_freq = exp2f(-(float)j * (13.2877123795494f / (float)half));
  const float ang = (float)n * inv_freq;
  float sn, cs;
  sincosf(ang, &sn, &cs);
  bf16* outs[4] = {qq, lq, qkk, lk};
  for (int h = 0; h < 4; ++h) {
    const float g1 = gamma[h * QK + j],        bb1 = beta[h * QK + j];
    const float g2 = gamma[h * QK + j + half], bb2 = beta[h * QK + j + half];
    const float h1 = a1 * g1 + bb1;
    const float h2 = a2 * g2 + bb2;
    bf16* o = outs[h] + (size_t)row * QK;
    o[j]        = (bf16)(h1 * cs - h2 * sn);
    o[j + half] = (bf16)(h2 * cs + h1 * sn);
  }
}

// ---------------- quad attention per (batch, group of 64) ----------------
// K tile (64x256 bf16, 32 KB) staged into LDS by the Tensor Data Mover.
__global__ void k_quad_attn(const bf16* __restrict__ qq, const bf16* __restrict__ qkk,
                            const bf16* __restrict__ vin, const float* __restrict__ pos_emb,
                            bf16* __restrict__ qo, int N, int H) {
  const int G = 64, QK = 256;
  const int g = blockIdx.x, b = blockIdx.y;
  const size_t rowBase = (size_t)b * N + (size_t)g * G;
  const bf16* Q  = qq  + rowBase * QK;
  const bf16* Km = qkk + rowBase * QK;
  const bf16* V  = vin + rowBase * H;
  __shared__ __align__(16) bf16 Ks[64][256];  // TDM-staged K tile (row n = B column n)
  __shared__ __align__(16) bf16 P[64][64];    // relu(S)^2 as bf16 (A operand of phase 2)
  __shared__ __align__(16) bf16 Vt[64][64];   // V tile transposed: (e_local, j)
  const int lane = threadIdx.x & 31;
  const int wid  = threadIdx.x >> 5;
  const int sm = (wid >> 1) << 4;             // 0,16,32,48
  const int sn = (wid & 1) << 5;              // 0,32
  // TDM DMA of the whole K tile; wave 0 issues, waits TENSORcnt, barrier releases all.
  if (wid == 0) {
    tdm_load_2d_bf16(Km, (uint32_t)(uintptr_t)&Ks[0][0], QK, G, QK);
    __builtin_amdgcn_s_wait_tensorcnt(0);
  }
  __syncthreads();
  // Phase 1: S = Q @ K^T over d=256 (A from global, B from LDS)
  v8f s0 = {}; v8f s1 = {};
  for (int k0 = 0; k0 < QK; k0 += 32) {
    v16bf a  = frag_a(Q + (size_t)sm * QK + k0, QK, lane);
    v16bf b0 = frag_b(&Ks[sn][k0], QK, lane);
    v16bf b1 = frag_b(&Ks[sn + 16][k0], QK, lane);
    s0 = wmma_bf16(a, b0, s0);
    s1 = wmma_bf16(a, b1, s1);
  }
  const int mo = (lane >> 4) << 3;
  const int c0 = sn + (lane & 15);
  const int c1 = c0 + 16;
  for (int r = 0; r < 8; ++r) {
    const int i = sm + r + mo;
    int d0 = i - c0; d0 = d0 < 0 ? -d0 : d0; if (d0 > 32) d0 = 32;
    int d1 = i - c1; d1 = d1 < 0 ? -d1 : d1; if (d1 > 32) d1 = 32;
    float e0 = s0[r] * (1.f / 64.f) + pos_emb[d0];
    float e1 = s1[r] * (1.f / 64.f) + pos_emb[d1];
    e0 = fmaxf(e0, 0.f); e1 = fmaxf(e1, 0.f);
    P[i][c0] = (bf16)(e0 * e0);
    P[i][c1] = (bf16)(e1 * e1);
  }
  // Phase 2: qo = P @ V, tiled over H in 64-wide slabs staged transposed in LDS
  for (int e0 = 0; e0 < H; e0 += 64) {
    __syncthreads();
    const int ec = threadIdx.x & 63;
    const int js = threadIdx.x >> 6;
    for (int jj = js; jj < G; jj += 4)
      Vt[ec][jj] = V[(size_t)jj * H + e0 + ec];
    __syncthreads();
    v8f o0 = {}; v8f o1 = {};
    for (int kk = 0; kk < G; kk += 32) {
      v16bf a  = frag_a(&P[sm][kk], 64, lane);
      v16bf b0 = frag_b(&Vt[sn][kk], 64, lane);
      v16bf b1 = frag_b(&Vt[sn + 16][kk], 64, lane);
      o0 = wmma_bf16(a, b0, o0);
      o1 = wmma_bf16(a, b1, o1);
    }
    for (int r = 0; r < 8; ++r) {
      const size_t orow = rowBase + (size_t)(sm + r + mo);
      qo[orow * H + e0 + c0] = (bf16)o0[r];
      qo[orow * H + e0 + c1] = (bf16)o1[r];
    }
  }
}

// ---------------- lin_kv^T[e][d] = sum_n v[n,e]*lk[n,d] / N (per batch) ----------------
__global__ void k_lin_kv(const bf16* __restrict__ lk, const bf16* __restrict__ vin,
                         bf16* __restrict__ kvT, int N, int QK, int H) {
  const int b = blockIdx.z;
  const int eBase = blockIdx.x * 64;
  const int dBase = blockIdx.y * 64;
  const bf16* Vb = vin + (size_t)b * N * H;
  const bf16* Kb = lk  + (size_t)b * N * QK;
  __shared__ __align__(16) bf16 As[64][32];  // (e_local, n_local) — transposed v chunk
  __shared__ __align__(16) bf16 Bs[64][32];  // (d_local, n_local) — transposed lk chunk
  const int lane = threadIdx.x & 31;
  const int wid  = threadIdx.x >> 5;
  const int sm = (wid >> 1) << 4;
  const int sn = (wid & 1) << 5;
  const int col  = threadIdx.x & 63;
  const int nset = threadIdx.x >> 6;         // 0..3
  v8f a0 = {}; v8f a1 = {};
  for (int n0 = 0; n0 < N; n0 += 32) {
    __syncthreads();
    for (int t = 0; t < 8; ++t) {
      const int nl = nset * 8 + t;
      As[col][nl] = Vb[(size_t)(n0 + nl) * H  + eBase + col];
      Bs[col][nl] = Kb[(size_t)(n0 + nl) * QK + dBase + col];
    }
    __syncthreads();
    v16bf a  = frag_a(&As[sm][0], 32, lane);
    v16bf b0 = frag_b(&Bs[sn][0], 32, lane);
    v16bf b1 = frag_b(&Bs[sn + 16][0], 32, lane);
    a0 = wmma_bf16(a, b0, a0);
    a1 = wmma_bf16(a, b1, a1);
  }
  const int mo = (lane >> 4) << 3;
  const int c0 = sn + (lane & 15);
  bf16* outB = kvT + (size_t)b * H * QK;
  const float sc = 1.f / (float)N;
  for (int r = 0; r < 8; ++r) {
    const int e = eBase + sm + r + mo;
    outB[(size_t)e * QK + dBase + c0]      = (bf16)(a0[r] * sc);
    outB[(size_t)e * QK + dBase + c0 + 16] = (bf16)(a1[r] * sc);
  }
}

// ---------------- lin_out = lq @ kv ; fused: act = gate * (qo + lin_out) in place ----------------
__global__ void k_lin_out_combine(const bf16* __restrict__ lq, const bf16* __restrict__ kvT,
                                  const bf16* __restrict__ gate, bf16* __restrict__ qo_act,
                                  int N, int QK, int H) {
  const int lane = threadIdx.x & 31;
  const int wid  = threadIdx.x >> 5;
  const int mBase = blockIdx.x * 64 + ((wid >> 1) << 4);
  const int nBase = blockIdx.y * 64 + ((wid & 1) << 5);
  const int b = (blockIdx.x * 64) / N;
  const bf16* Arow = lq + (size_t)mBase * QK;
  const bf16* Wb = kvT + (size_t)b * H * QK;
  const bf16* W0 = Wb + (size_t)nBase * QK;
  const bf16* W1 = Wb + (size_t)(nBase + 16) * QK;
  v8f acc0 = {}; v8f acc1 = {};
  for (int k0 = 0; k0 < QK; k0 += 32) {
    v16bf a  = frag_a(Arow + k0, QK, lane);
    v16bf b0 = frag_b(W0 + k0, QK, lane);
    v16bf b1 = frag_b(W1 + k0, QK, lane);
    acc0 = wmma_bf16(a, b0, acc0);
    acc1 = wmma_bf16(a, b1, acc1);
  }
  const int mo = (lane >> 4) << 3;
  const int c0 = nBase + (lane & 15);
  for (int r = 0; r < 8; ++r) {
    const size_t row = (size_t)(mBase + r + mo);
    const size_t i0 = row * H + c0;
    const size_t i1 = i0 + 16;
    qo_act[i0] = (bf16)((float)gate[i0] * ((float)qo_act[i0] + acc0[r]));
    qo_act[i1] = (bf16)((float)gate[i1] * ((float)qo_act[i1] + acc1[r]));
  }
}

extern "C" void kernel_launch(void* const* d_in, const int* in_sizes, int n_in,
                              void* d_out, int out_size, void* d_ws, size_t ws_size,
                              hipStream_t stream) {
  (void)in_sizes; (void)n_in; (void)out_size; (void)ws_size;
  const int B = 4, N = 4096, D = 1024, QK = 256, H = 2048;
  const int M = B * N;  // 16384
  const float* x     = (const float*)d_in[0];
  const float* ln1w  = (const float*)d_in[1];
  const float* ln1b  = (const float*)d_in[2];
  const float* Wqk   = (const float*)d_in[3];
  const float* bqk   = (const float*)d_in[4];
  const float* Wv    = (const float*)d_in[5];
  const float* bv    = (const float*)d_in[6];
  const float* Wg    = (const float*)d_in[7];
  const float* bg    = (const float*)d_in[8];
  const float* Wo    = (const float*)d_in[9];
  const float* bo    = (const float*)d_in[10];
  const float* ln2w  = (const float*)d_in[11];
  const float* ln2b  = (const float*)d_in[12];
  const float* gamma = (const float*)d_in[13];
  const float* beta  = (const float*)d_in[14];
  const float* pe    = (const float*)d_in[15];

  char* ws = (char*)d_ws;
  size_t off = 0;
  auto alloc = [&](size_t bytes) -> void* {
    void* p = ws + off;
    off = (off + bytes + 255) & ~(size_t)255;
    return p;
  };
  bf16* xln  = (bf16*)alloc((size_t)M * D * 2);
  bf16* wqk  = (bf16*)alloc((size_t)QK * D * 2);
  bf16* wv   = (bf16*)alloc((size_t)H * D * 2);
  bf16* wg   = (bf16*)alloc((size_t)H * D * 2);
  bf16* wo   = (bf16*)alloc((size_t)D * H * 2);
  bf16* qkp  = (bf16*)alloc((size_t)M * QK * 2);
  bf16* vact = (bf16*)alloc((size_t)M * H * 2);
  bf16* gate = (bf16*)alloc((size_t)M * H * 2);
  bf16* qq   = (bf16*)alloc((size_t)M * QK * 2);
  bf16* lqh  = (bf16*)alloc((size_t)M * QK * 2);
  bf16* qkk  = (bf16*)alloc((size_t)M * QK * 2);
  bf16* lkh  = (bf16*)alloc((size_t)M * QK * 2);
  bf16* qo   = (bf16*)alloc((size_t)M * H * 2);
  bf16* kvT  = (bf16*)alloc((size_t)B * H * QK * 2);
  float* pre = (float*)vact;  // reuse: vact (M*H*2 bytes) dead after k_lin_kv == M*D*4 bytes

  // 1. LN1 -> bf16 activations
  k_layernorm<bf16><<<M, 256, 0, stream>>>(x, ln1w, ln1b, xln, D);
  // 2. weights -> bf16
  k_cvt_bf16<<<(QK * D + 255) / 256, 256, 0, stream>>>(Wqk, wqk, QK * D);
  k_cvt_bf16<<<(H * D + 255) / 256, 256, 0, stream>>>(Wv, wv, H * D);
  k_cvt_bf16<<<(H * D + 255) / 256, 256, 0, stream>>>(Wg, wg, H * D);
  k_cvt_bf16<<<(D * H + 255) / 256, 256, 0, stream>>>(Wo, wo, D * H);
  // 3. projections with fused bias+SiLU (WMMA, 64x128 block tiles)
  k_gemm<true, bf16><<<dim3(M / 64, QK / 128), 256, 0, stream>>>(xln, wqk, bqk, qkp, D, QK);
  k_gemm<true, bf16><<<dim3(M / 64, H / 128), 256, 0, stream>>>(xln, wv, bv, vact, D, H);
  k_gemm<true, bf16><<<dim3(M / 64, H / 128), 256, 0, stream>>>(xln, wg, bg, gate, D, H);
  // 4. per-head scale/offset + RoPE
  k_heads_rope<<<M, 128, 0, stream>>>(qkp, gamma, beta, qq, lqh, qkk, lkh, N, QK);
  // 5. quadratic (in-group) attention (WMMA x2, TDM-staged K tile)
  k_quad_attn<<<dim3(N / 64, B), 256, 0, stream>>>(qq, qkk, vact, pe, qo, N, H);
  // 6. linear attention state kv^T (WMMA, LDS-transposed operands)
  k_lin_kv<<<dim3(H / 64, QK / 64, B), 256, 0, stream>>>(lkh, vact, kvT, N, QK, H);
  // 7. linear attention out + fused gate*(qo+lo) in place (WMMA)
  k_lin_out_combine<<<dim3(M / 64, H / 64), 256, 0, stream>>>(lqh, kvT, gate, qo, N, QK, H);
  // 8. output projection (WMMA, f32 out)
  k_gemm<false, float><<<dim3(M / 64, D / 128), 256, 0, stream>>>(qo, wo, bo, pre, H, D);
  // 9. LN2 -> d_out (f32)
  k_layernorm<float><<<M, 256, 0, stream>>>(pre, ln2w, ln2b, (float*)d_out, D);
}
